// BioSleepXWithMRCNN_90237262889250
// MI455X (gfx1250) — matrix operand order repaired and checked by reading, not compiled, not run on hardware
//
#include <hip/hip_runtime.h>
#include <hip/hip_bf16.h>

typedef _Float16 v8h  __attribute__((ext_vector_type(8)));
typedef _Float16 v16h __attribute__((ext_vector_type(16)));
typedef float    v8f  __attribute__((ext_vector_type(8)));

#define RS_BN 0.9999950000374997f   /* 1/sqrt(1+1e-5) */
#define NB 256                       /* batch */

/* ---------------- signal normalize: (x-mu)/(std_ddof1 + 1e-6) ---------------- */
__global__ void normalize_k(const float* __restrict__ x, float* __restrict__ xn) {
  int b = blockIdx.x, t = threadIdx.x;
  __shared__ float rs[256], rq[256];
  const float* xr = x + (size_t)b * 3000;
  float s = 0.f, q = 0.f;
  for (int i = t; i < 3000; i += 256) { float v = xr[i]; s += v; q += v * v; }
  rs[t] = s; rq[t] = q; __syncthreads();
  for (int sh = 128; sh > 0; sh >>= 1) {
    if (t < sh) { rs[t] += rs[t + sh]; rq[t] += rq[t + sh]; }
    __syncthreads();
  }
  float mu  = rs[0] / 3000.f;
  float var = (rq[0] - 3000.f * mu * mu) / 2999.f;
  float inv = 1.f / (sqrtf(fmaxf(var, 0.f)) + 1e-6f);
  float* o = xn + (size_t)b * 3000;
  for (int i = t; i < 3000; i += 256) o[i] = (xr[i] - mu) * inv;
}

/* ---------------- conv1 (Cin=1) + BN + ReLU, LDS-staged window ---------------- */
__global__ void conv1_k(const float* __restrict__ xn, const float* __restrict__ w,
                        const float* __restrict__ g, const float* __restrict__ bb,
                        float* __restrict__ out, int Lout, int KW, int stride, int pad) {
  int l = blockIdx.x, b = blockIdx.y, c = threadIdx.x;   // 64 threads (channels)
  extern __shared__ float sx[];
  int base = l * stride - pad;
  for (int i = c; i < KW; i += 64) {
    int p = base + i;
    sx[i] = (p >= 0 && p < 3000) ? xn[(size_t)b * 3000 + p] : 0.f;
  }
  __syncthreads();
  const float* wr = w + (size_t)c * KW;
  float s = 0.f;
  for (int k = 0; k < KW; ++k) s += sx[k] * wr[k];
  float d = fmaxf(s * (g[c] * RS_BN) + bb[c], 0.f);
  out[((size_t)b * 64 + c) * Lout + l] = d;
}

/* ---------------- maxpool -> padded f16 buffer (width 40) for implicit GEMM ---------------- */
__global__ void pool_k(const float* __restrict__ cin, _Float16* __restrict__ buf,
                       int Lin, int k, int stride, int ppad, int Lvalid, int padL) {
  int idx = blockIdx.x * 256 + threadIdx.x;
  int total = NB * 64 * 40;
  if (idx >= total) return;
  int w = idx % 40; int bc = idx / 40;
  int pos = w - padL;
  float m = 0.f;
  if (pos >= 0 && pos < Lvalid) {
    int st = pos * stride - ppad;
    m = -1e30f;
    const float* row = cin + (size_t)bc * Lin;
    for (int i = 0; i < k; ++i) { int q = st + i; if (q >= 0 && q < Lin) m = fmaxf(m, row[q]); }
  }
  buf[idx] = (_Float16)m;
}

/* ---------------- conv2 as implicit GEMM via WMMA f16 ----------------
   One wave = 16 output columns x ALL 128 output channels (8 register-blocked
   accumulator tiles). The expensive per-column patch gather (B fragment) is
   built once per K-chunk and reused across all 8 Cout tiles. */
template<int KW>
__global__ void conv2_wmma_k(const _Float16* __restrict__ Wc, const _Float16* __restrict__ buf,
                             const float* __restrict__ g, const float* __restrict__ bb,
                             float* __restrict__ xc, int Ktot, int co_off) {
  int lane = threadIdx.x;
  int h = lane >> 4, r = lane & 15;
  int col = blockIdx.x * 16 + r;           // column = b*32 + l
  int b = col >> 5, l = col & 31;
  const _Float16* bbase = buf + (size_t)b * 64 * 40;
  v8f acc[8];
  #pragma unroll
  for (int mt = 0; mt < 8; ++mt)
    #pragma unroll
    for (int i = 0; i < 8; ++i) acc[mt][i] = 0.f;
  for (int kc = 0; kc < Ktot; kc += 32) {
    v16h bf;
    int kbase = kc + 8 * h;
    #pragma unroll
    for (int i = 0; i < 16; ++i) {
      int k  = kbase + i + ((i >> 3) << 3);
      int ci = k / KW;
      int kt = k - ci * KW;
      bf[i] = bbase[ci * 40 + l + kt];     // padded buffer: idx = l + kt
    }
    #pragma unroll
    for (int mt = 0; mt < 8; ++mt) {
      const _Float16* Arow = Wc + (size_t)(mt * 16 + r) * Ktot + kc;
      v8h a0 = *(const v8h*)(Arow + 8 * h);
      v8h a1 = *(const v8h*)(Arow + 16 + 8 * h);
      v16h a;
      #pragma unroll
      for (int i = 0; i < 8; ++i) { a[i] = a0[i]; a[8 + i] = a1[i]; }
      acc[mt] = __builtin_amdgcn_wmma_f32_16x16x32_f16(false, a, false, bf, (short)0, acc[mt], false, false);
    }
  }
  #pragma unroll
  for (int mt = 0; mt < 8; ++mt) {
    #pragma unroll
    for (int v = 0; v < 8; ++v) {
      int co = mt * 16 + v + 8 * h;
      float d = fmaxf(acc[mt][v] * (g[co] * RS_BN) + bb[co], 0.f);
      xc[((size_t)b * 256 + co_off + co) * 32 + l] = d;
    }
  }
}

/* ---------------- SE block: squeeze, 256->32->256 MLP, sigmoid scale -> fused ---------------- */
__global__ void se_k(const float* __restrict__ xc, const float* __restrict__ w1,
                     const float* __restrict__ w2, float* __restrict__ fused, int ch_off) {
  int b = blockIdx.x, t = threadIdx.x;     // 256 threads
  __shared__ float mv[256], z[32];
  const float* row = xc + ((size_t)b * 256 + t) * 32;
  float s = 0.f;
  for (int l = 0; l < 32; ++l) s += row[l];
  mv[t] = s * (1.f / 32.f);
  __syncthreads();
  if (t < 32) {
    float u = 0.f;
    for (int c = 0; c < 256; ++c) u += mv[c] * w1[t * 256 + c];
    z[t] = fmaxf(u, 0.f);
  }
  __syncthreads();
  float u = 0.f;
  for (int j = 0; j < 32; ++j) u += z[j] * w2[t * 32 + j];
  float y = 1.f / (1.f + expf(-u));
  float* orow = fused + ((size_t)b * 512 + ch_off + t) * 32;
  for (int l = 0; l < 32; ++l) orow[l] = row[l] * y;
}

/* ---------------- fuse 1x1 conv 512->30 + BN + ReLU ---------------- */
__global__ void fuse1_k(const float* __restrict__ fused, const float* __restrict__ w,
                        const float* __restrict__ bias, const float* __restrict__ g,
                        const float* __restrict__ bb, float* __restrict__ x1) {
  int b = blockIdx.x; int l = threadIdx.x; int o = threadIdx.y;   // (32,30)
  const float* f  = fused + (size_t)b * 512 * 32 + l;
  const float* wr = w + (size_t)o * 512;
  float s = 0.f;
  for (int c = 0; c < 512; ++c) s += f[(size_t)c * 32] * wr[c];
  float d = fmaxf((s + bias[o]) * (g[o] * RS_BN) + bb[o], 0.f);
  x1[((size_t)b * 30 + o) * 32 + l] = d;
}

/* ---------------- adj 1x1 conv 30->80 + BN + ReLU ---------------- */
__global__ void adj_k(const float* __restrict__ x1, const float* __restrict__ w,
                      const float* __restrict__ bias, const float* __restrict__ g,
                      const float* __restrict__ bb, float* __restrict__ x2) {
  int b = blockIdx.x; int l = threadIdx.x; int o = blockIdx.y * 16 + threadIdx.y;
  const float* f  = x1 + (size_t)b * 30 * 32 + l;
  const float* wr = w + (size_t)o * 30;
  float s = 0.f;
  for (int c = 0; c < 30; ++c) s += f[(size_t)c * 32] * wr[c];
  float d = fmaxf((s + bias[o]) * (g[o] * RS_BN) + bb[o], 0.f);
  x2[((size_t)b * 80 + o) * 32 + l] = d;
}

/* ---------------- adaptive avg pool 32->80 + transpose to tokens ---------------- */
__global__ void adapt_k(const float* __restrict__ x2, float* __restrict__ xtok) {
  int idx = blockIdx.x * 256 + threadIdx.x;
  if (idx >= NB * 80 * 80) return;
  int c = idx % 80; int rem = idx / 80; int i = rem % 80; int b = rem / 80;
  int s = (i * 32) / 80;
  int e = ((i + 1) * 32 + 79) / 80;
  float a = 0.f;
  for (int l = s; l < e; ++l) a += x2[((size_t)b * 80 + c) * 32 + l];
  xtok[idx] = a / (float)(e - s);
}

/* ---------------- custom LayerNorm (std ddof=1, eps added to sd) ---------------- */
__global__ void ln_k(const float* __restrict__ X, const float* __restrict__ g,
                     const float* __restrict__ bb, float* __restrict__ o32,
                     _Float16* __restrict__ o16, int Kp) {
  int row = blockIdx.x, t = threadIdx.x;   // 128 threads
  __shared__ float red[128];
  __shared__ float smean, sinv;
  float v = (t < 80) ? X[(size_t)row * 80 + t] : 0.f;
  red[t] = v; __syncthreads();
  for (int s = 64; s > 0; s >>= 1) { if (t < s) red[t] += red[t + s]; __syncthreads(); }
  if (t == 0) smean = red[0] * (1.f / 80.f);
  __syncthreads();
  float mu = smean;
  float dv = (t < 80) ? v - mu : 0.f;
  red[t] = dv * dv; __syncthreads();
  for (int s = 64; s > 0; s >>= 1) { if (t < s) red[t] += red[t + s]; __syncthreads(); }
  if (t == 0) sinv = 1.f / (sqrtf(red[0] * (1.f / 79.f)) + 1e-6f);
  __syncthreads();
  if (t < 80) {
    float o = g[t] * dv * sinv + bb[t];
    if (o32) o32[(size_t)row * 80 + t] = o;
    if (o16) o16[(size_t)row * Kp + t] = (_Float16)o;
  } else if (o16 && t < Kp) {
    o16[(size_t)row * Kp + t] = (_Float16)0.f;   // zero K-padding
  }
}

/* ---------------- generic WMMA GEMM, register-blocked over all N tiles ----------------
   One wave = 16 rows of A x all NT*16 columns. A fragment loaded once per K-chunk
   and reused for NT independent WMMAs. */
template<int NT>
__global__ void gemm_k(const _Float16* __restrict__ A, const _Float16* __restrict__ W,
                       const float* __restrict__ bias, int Nvalid,
                       float* __restrict__ O32, int ldo32, int accum,
                       _Float16* __restrict__ O16, int ldo16,
                       int Kp, int relu) {
  int lane = threadIdx.x;
  int h = lane >> 4, r = lane & 15;
  int mtile = blockIdx.x;
  const _Float16* Arow = A + (size_t)(mtile * 16 + r) * Kp;
  v8f acc[NT];
  #pragma unroll
  for (int nt = 0; nt < NT; ++nt)
    #pragma unroll
    for (int i = 0; i < 8; ++i) acc[nt][i] = 0.f;
  for (int kc = 0; kc < Kp; kc += 32) {
    v8h a0 = *(const v8h*)(Arow + kc + 16 * h);
    v8h a1 = *(const v8h*)(Arow + kc + 16 * h + 8);
    v16h a;
    #pragma unroll
    for (int i = 0; i < 8; ++i) { a[i] = a0[i]; a[8 + i] = a1[i]; }
    #pragma unroll
    for (int nt = 0; nt < NT; ++nt) {
      v16h bf = *(const v16h*)(W + (size_t)(nt * 16 + r) * Kp + kc + 16 * h);
      acc[nt] = __builtin_amdgcn_wmma_f32_16x16x32_f16(false, a, false, bf, (short)0, acc[nt], false, false);
    }
  }
  #pragma unroll
  for (int nt = 0; nt < NT; ++nt) {
    int n = nt * 16 + r;
    float bv = (bias && n < Nvalid) ? bias[n] : 0.f;
    #pragma unroll
    for (int v = 0; v < 8; ++v) {
      int m = mtile * 16 + v + 8 * h;
      float d = acc[nt][v] + bv;
      if (relu) d = fmaxf(d, 0.f);
      if (O32) {
        if (accum) O32[(size_t)m * ldo32 + n] += d;
        else       O32[(size_t)m * ldo32 + n]  = d;
      }
      if (O16) O16[(size_t)m * ldo16 + n] = (_Float16)d;
    }
  }
}

/* ---------------- additive attention: scores = sum_d tanh(Q_i+K_j)*w_d + c0; softmax; aV ---------------- */
__global__ void attn_k(const float* __restrict__ Q, const float* __restrict__ K,
                       const float* __restrict__ V,
                       const float* __restrict__ dw, const float* __restrict__ sw,
                       const float* __restrict__ tw, const float* __restrict__ db,
                       const float* __restrict__ sb, const float* __restrict__ tb,
                       float* __restrict__ X) {
  int row = blockIdx.x; int b = row / 80; int t = threadIdx.x;   // 80 threads
  __shared__ float q[80], wv[80], av[80], red[128];
  __shared__ float sm, ssum;
  q[t]  = Q[(size_t)row * 80 + t];
  wv[t] = 0.1118033988749895f + dw[t] + sw[t] + tw[t];           // 1/sqrt(80) + dw+sw+tw
  __syncthreads();
  const float* Kb = K + (size_t)b * 6400 + (size_t)t * 80;       // t plays role of j
  float s = db[0] + sb[0] + tb[0];
  for (int d = 0; d < 80; ++d) s += tanhf(q[d] + Kb[d]) * wv[d];
  red[t] = s; if (t < 48) red[t + 80] = -1e30f;
  __syncthreads();
  for (int sh = 64; sh > 0; sh >>= 1) { if (t < sh) red[t] = fmaxf(red[t], red[t + sh]); __syncthreads(); }
  if (t == 0) sm = red[0];
  __syncthreads();
  float e = expf(s - sm);
  av[t] = e;
  red[t] = e; if (t < 48) red[t + 80] = 0.f;
  __syncthreads();
  for (int sh = 64; sh > 0; sh >>= 1) { if (t < sh) red[t] += red[t + sh]; __syncthreads(); }
  if (t == 0) ssum = red[0];
  __syncthreads();
  const float* Vb = V + (size_t)b * 6400;
  float o = 0.f;
  for (int j = 0; j < 80; ++j) o += av[j] * Vb[(size_t)j * 80 + t];   // t plays role of c
  X[(size_t)row * 80 + t] += o / ssum;                                // residual add
}

/* ---------------- head: token-mean -> 80->40 relu -> 40->5 ---------------- */
__global__ void head_k(const float* __restrict__ xf, const float* __restrict__ cw1,
                       const float* __restrict__ cb1, const float* __restrict__ cw2,
                       const float* __restrict__ cb2, float* __restrict__ out) {
  int b = blockIdx.x, t = threadIdx.x;   // 80 threads
  __shared__ float feat[80], hh[40];
  float s = 0.f;
  for (int i = 0; i < 80; ++i) s += xf[((size_t)b * 80 + i) * 80 + t];
  feat[t] = s * (1.f / 80.f);
  __syncthreads();
  if (t < 40) {
    float u = cb1[t];
    for (int c = 0; c < 80; ++c) u += feat[c] * cw1[t * 80 + c];
    hh[t] = fmaxf(u, 0.f);
  }
  __syncthreads();
  if (t < 5) {
    float u = cb2[t];
    for (int j = 0; j < 40; ++j) u += hh[j] * cw2[t * 40 + j];
    out[(size_t)b * 5 + t] = u;
  }
}

/* ---------------- f32 -> f16 weight convert with K/N zero padding ---------------- */
__global__ void cvtpad_k(const float* __restrict__ src, _Float16* __restrict__ dst,
                         int R, int Kc, int Kp, int total) {
  int i = blockIdx.x * 256 + threadIdx.x;
  if (i >= total) return;
  int r = i / Kp, k = i - r * Kp;
  dst[i] = (r < R && k < Kc) ? (_Float16)src[(size_t)r * Kc + k] : (_Float16)0.f;
}

extern "C" void kernel_launch(void* const* d_in, const int* in_sizes, int n_in,
                              void* d_out, int out_size, void* d_ws, size_t ws_size,
                              hipStream_t stream) {
  (void)in_sizes; (void)n_in; (void)out_size; (void)ws_size;
  const float* x_sig[2] = { (const float*)d_in[0], (const float*)d_in[1] };
  auto P = [&](int i) { return (const float*)d_in[i]; };
  // param index map: eeg 2..15, eog 16..29, fuse 30..33, adj 34..37,
  // layers 38..57 / 58..77, final 78..79, head 80..83

  float* ws    = (float*)d_ws;
  float* fused = ws;                       // 256*512*32
  float* xtok  = fused + 4194304;          // 256*80*80
  float* scr   = xtok + 1638400;           // reused scratch (7,208,960 floats)
  _Float16* wpool = (_Float16*)(scr + 7208960);

  // per-signal scratch layout
  float*    xn   = scr;
  float*    c1a  = scr + 768000;
  float*    c1b  = scr + 2078720;
  float*    xcb  = scr + 3078144;
  _Float16* buf1 = (_Float16*)(scr + 5175296);
  _Float16* buf2 = (_Float16*)(scr + 5502976);
  _Float16* wc1  = wpool;                  // 128x512
  _Float16* wc2  = wpool + 65536;          // 128x384

  for (int s = 0; s < 2; ++s) {
    int pb = 2 + s * 14;
    cvtpad_k<<<(65536 + 255) / 256, 256, 0, stream>>>(P(pb + 3), wc1, 128, 512, 512, 65536);
    cvtpad_k<<<(49152 + 255) / 256, 256, 0, stream>>>(P(pb + 9), wc2, 128, 384, 384, 49152);
    normalize_k<<<NB, 256, 0, stream>>>(x_sig[s], xn);
    // path1: only first 80 conv positions are ever needed (m = min(252,32) = 32 downstream)
    conv1_k<<<dim3(80, NB), 64, 50 * 4, stream>>>(xn, P(pb + 0), P(pb + 1), P(pb + 2), c1a, 80, 50, 6, 24);
    conv1_k<<<dim3(61, NB), 64, 400 * 4, stream>>>(xn, P(pb + 6), P(pb + 7), P(pb + 8), c1b, 61, 400, 50, 200);
    pool_k<<<(NB * 64 * 40 + 255) / 256, 256, 0, stream>>>(c1a, buf1, 80, 8, 2, 4, 36, 4);
    pool_k<<<(NB * 64 * 40 + 255) / 256, 256, 0, stream>>>(c1b, buf2, 61, 4, 2, 2, 31, 3);
    conv2_wmma_k<8><<<512, 32, 0, stream>>>(wc1, buf1, P(pb + 4), P(pb + 5), xcb, 512, 0);
    conv2_wmma_k<6><<<512, 32, 0, stream>>>(wc2, buf2, P(pb + 10), P(pb + 11), xcb, 384, 128);
    se_k<<<NB, 256, 0, stream>>>(xcb, P(pb + 12), P(pb + 13), fused, s * 256);
  }

  // fuse -> adj -> adaptive avg -> tokens
  float* x1 = scr;
  float* x2 = scr + 245760;
  fuse1_k<<<NB, dim3(32, 30), 0, stream>>>(fused, P(30), P(31), P(32), P(33), x1);
  adj_k<<<dim3(NB, 5), dim3(32, 16), 0, stream>>>(x1, P(34), P(35), P(36), P(37), x2);
  adapt_k<<<(NB * 80 * 80 + 255) / 256, 256, 0, stream>>>(x2, xtok);

  // transformer weights -> f16 (K padded)
  for (int L = 0; L < 2; ++L) {
    int lb = 38 + L * 20;
    _Float16* wb = wpool + 114688 + L * 45568;
    cvtpad_k<<<(80 * 96 + 255) / 256, 256, 0, stream>>>(P(lb + 0),  wb + 0,     80, 80, 96, 80 * 96);
    cvtpad_k<<<(80 * 96 + 255) / 256, 256, 0, stream>>>(P(lb + 2),  wb + 7680,  80, 80, 96, 80 * 96);
    cvtpad_k<<<(80 * 96 + 255) / 256, 256, 0, stream>>>(P(lb + 4),  wb + 15360, 80, 80, 96, 80 * 96);
    cvtpad_k<<<(128 * 96 + 255) / 256, 256, 0, stream>>>(P(lb + 16), wb + 23040, 120, 80, 96, 128 * 96);
    cvtpad_k<<<(80 * 128 + 255) / 256, 256, 0, stream>>>(P(lb + 18), wb + 35328, 80, 120, 128, 80 * 128);
  }

  _Float16* h16 = (_Float16*)scr;
  float* Q  = scr + 983040;
  float* Km = scr + 2621440;
  float* V  = scr + 4259840;
  _Float16* u16 = (_Float16*)(scr + 5898240);
  const int M = NB * 80, Mt = M / 16;   // 20480 rows, 1280 tiles

  for (int L = 0; L < 2; ++L) {
    int lb = 38 + L * 20;
    _Float16* wb = wpool + 114688 + L * 45568;
    ln_k<<<M, 128, 0, stream>>>(xtok, P(lb + 12), P(lb + 13), nullptr, h16, 96);
    gemm_k<5><<<Mt, 32, 0, stream>>>(h16, wb + 0,     P(lb + 1), 80,  Q,  80, 0, nullptr, 0,   96, 0);
    gemm_k<5><<<Mt, 32, 0, stream>>>(h16, wb + 7680,  P(lb + 3), 80,  Km, 80, 0, nullptr, 0,   96, 0);
    gemm_k<5><<<Mt, 32, 0, stream>>>(h16, wb + 15360, P(lb + 5), 80,  V,  80, 0, nullptr, 0,   96, 0);
    attn_k<<<M, 80, 0, stream>>>(Q, Km, V, P(lb + 6), P(lb + 8), P(lb + 10),
                                 P(lb + 7), P(lb + 9), P(lb + 11), xtok);
    ln_k<<<M, 128, 0, stream>>>(xtok, P(lb + 14), P(lb + 15), nullptr, h16, 96);
    gemm_k<8><<<Mt, 32, 0, stream>>>(h16, wb + 23040, P(lb + 17), 120, nullptr, 0, 0, u16, 128, 96, 1);
    gemm_k<5><<<Mt, 32, 0, stream>>>(u16, wb + 35328, P(lb + 19), 80,  xtok, 80, 1, nullptr, 0, 128, 0);
  }

  // final LN -> mean over tokens -> classifier
  float* xf = scr;   // scratch free again
  ln_k<<<M, 128, 0, stream>>>(xtok, P(78), P(79), xf, nullptr, 0);
  head_k<<<NB, 80, 0, stream>>>(xf, P(80), P(81), P(82), P(83), (float*)d_out);
}
